// Qwen2LMInpaint_25735444037998
// MI455X (gfx1250) — matrix operand order repaired and checked by reading, not compiled, not run on hardware
//
#include <hip/hip_runtime.h>
#include <hip/hip_bf16.h>
#include <math.h>

// Problem constants (from reference): B=8, L=2048, D=896, V=512
#define BB   8
#define LL   2048
#define DD   896
#define VV   512
#define MM   (BB * LL)        // 16384 tokens
#define K1   (3 * DD)         // 2688
#define K2   (DD)             // 896
#define HPAD (DD + 8)         // LDS row stride in bf16: 452 dwords == 4 mod 64 banks

typedef __attribute__((ext_vector_type(8)))  __bf16 v8bf;
typedef __attribute__((ext_vector_type(16))) __bf16 v16bf;
typedef __attribute__((ext_vector_type(8)))  float  v8f;

__device__ __forceinline__ v16bf make_frag(const __bf16* p0, const __bf16* p1) {
    v8bf lo = *(const v8bf*)p0;
    v8bf hi = *(const v8bf*)p1;
    return __builtin_shufflevector(lo, hi, 0,1,2,3,4,5,6,7,8,9,10,11,12,13,14,15);
}

// ---------------- prep kernels ----------------

__global__ void cvt_bf16_kernel(const float* __restrict__ src, __bf16* __restrict__ dst, int n) {
    int i = blockIdx.x * blockDim.x + threadIdx.x;
    if (i < n) dst[i] = (__bf16)src[i];
}

// W[k*N + n] (row-major K x N) -> Wt[n*K + k] (bf16, K contiguous)
__global__ void transpose_cvt_kernel(const float* __restrict__ W, __bf16* __restrict__ Wt,
                                     int K, int N) {
    int i = blockIdx.x * blockDim.x + threadIdx.x;
    if (i < K * N) {
        int k = i / N;
        int n = i - k * N;
        Wt[(size_t)n * K + k] = (__bf16)W[i];
    }
}

// per-token: masked token ids (4 per token) + pf_mask output (as float 0/1)
__global__ void tok_mask_kernel(const int* __restrict__ pf, const int* __restrict__ len,
                                int* __restrict__ tok, float* __restrict__ mask_out) {
    int i = blockIdx.x * blockDim.x + threadIdx.x;   // token index in [0, MM)
    if (i >= MM) return;
    int b   = i / LL;
    int pos = i - b * LL;
    bool valid = pos < len[b];
    bool any = false;
    #pragma unroll
    for (int j = 0; j < 4; ++j) {
        int t = valid ? pf[(size_t)b * (4 * LL) + 4 * pos + j] : 0;
        tok[i * 4 + j] = t;
        any |= (t != 0);
    }
    mask_out[i] = any ? 1.0f : 0.0f;
}

// ---------------- fused MLP kernel ----------------
// grid.x = MM/32 blocks, 256 threads (8 waves). Block = 32 M-rows x full N=896.
// Wave w: N in [w*112, w*112+112), M-tiles {m0, m0+16}; 14 accumulators.
// Phase 1: h = gelu(cat @ W1 + b1) -> LDS tile (32 x HPAD bf16)
// Phase 2: out = h @ W2 + b2 + table[to]   (A from LDS, no global round-trip)
__global__ __launch_bounds__(256) void fused_mlp_kernel(
    const __bf16* __restrict__ tableb,   // [VV][DD] bf16
    const __bf16* __restrict__ W1t,      // [n][K1] bf16 (transposed)
    const __bf16* __restrict__ W2t,      // [n][K2] bf16 (transposed)
    const int*    __restrict__ tok,      // [MM][4]
    const float*  __restrict__ b1,       // [DD]
    const float*  __restrict__ b2,       // [DD]
    const float*  __restrict__ table32,  // [VV][DD] fp32 residual ("to")
    float*        __restrict__ out)      // [MM][DD] fp32
{
    __shared__ __bf16 hs[32 * HPAD];     // 57,856 B

    const int m0   = blockIdx.x * 32;
    const int wave = threadIdx.x >> 5;
    const int lane = threadIdx.x & 31;
    const int ml   = lane & 15;
    const int kg   = lane >> 4;
    const int n0   = wave * 112;

    // -------- phase 1: gather + GEMM1 + GELU -> LDS --------
    {
        const __bf16* aLo[3];
        const __bf16* aHi[3];
        {
            const int rLo = m0 + ml, rHi = m0 + 16 + ml;
            #pragma unroll
            for (int j = 0; j < 3; ++j) {
                aLo[j] = tableb + (size_t)tok[rLo * 4 + j] * DD + kg * 8;
                aHi[j] = tableb + (size_t)tok[rHi * 4 + j] * DD + kg * 8;
            }
        }
        const __bf16* wb = W1t + (size_t)(n0 + ml) * K1 + kg * 8;

        v8f acc[14] = {};

        #pragma unroll
        for (int seg = 0; seg < 3; ++seg) {      // fixed segments: no runtime division
            const __bf16* apLo = aLo[seg];
            const __bf16* apHi = aHi[seg];
            const __bf16* bp   = wb + seg * DD;
            for (int kk = 0; kk < DD; kk += 32) {
                v16bf a0 = make_frag(apLo, apLo + 16);
                v16bf a1 = make_frag(apHi, apHi + 16);
                v16bf bf[7];
                #pragma unroll
                for (int t = 0; t < 7; ++t) {
                    const __bf16* p = bp + (size_t)t * 16 * K1;   // imm offsets
                    bf[t] = make_frag(p, p + 16);
                }
                #pragma unroll
                for (int t = 0; t < 7; ++t) {
                    acc[2 * t + 0] = __builtin_amdgcn_wmma_f32_16x16x32_bf16(
                        false, a0, false, bf[t], (short)0, acc[2 * t + 0], false, false);
                    acc[2 * t + 1] = __builtin_amdgcn_wmma_f32_16x16x32_bf16(
                        false, a1, false, bf[t], (short)0, acc[2 * t + 1], false, false);
                }
                apLo += 32; apHi += 32; bp += 32;
            }
        }

        // epilogue 1: bias + exact GELU, write bf16 tile to LDS
        // element (vgpr i, lane) -> local row = mt*16 + i + 8*kg, col n = n0 + t*16 + ml
        #pragma unroll
        for (int t = 0; t < 7; ++t) {
            const int n = n0 + t * 16 + ml;
            const float bias = b1[n];
            #pragma unroll
            for (int mt = 0; mt < 2; ++mt) {
                #pragma unroll
                for (int i = 0; i < 8; ++i) {
                    const int lr = mt * 16 + i + 8 * kg;
                    float x = acc[2 * t + mt][i] + bias;
                    float g = 0.5f * x * (1.0f + erff(x * 0.70710678118654752f));
                    hs[lr * HPAD + n] = (__bf16)g;
                }
            }
        }
    }

    __syncthreads();

    // -------- phase 2: GEMM2 from LDS + bias + residual -> out --------
    {
        const __bf16* bp = W2t + (size_t)(n0 + ml) * K2 + kg * 8;

        v8f acc[14] = {};

        const __bf16* apLo = &hs[ml * HPAD + kg * 8];
        const __bf16* apHi = &hs[(16 + ml) * HPAD + kg * 8];

        for (int kk = 0; kk < DD; kk += 32) {
            v16bf a0 = make_frag(apLo, apLo + 16);     // ds_load_b128 x2
            v16bf a1 = make_frag(apHi, apHi + 16);
            v16bf bf[7];
            #pragma unroll
            for (int t = 0; t < 7; ++t) {
                const __bf16* p = bp + (size_t)t * 16 * K2;
                bf[t] = make_frag(p, p + 16);
            }
            #pragma unroll
            for (int t = 0; t < 7; ++t) {
                acc[2 * t + 0] = __builtin_amdgcn_wmma_f32_16x16x32_bf16(
                    false, a0, false, bf[t], (short)0, acc[2 * t + 0], false, false);
                acc[2 * t + 1] = __builtin_amdgcn_wmma_f32_16x16x32_bf16(
                    false, a1, false, bf[t], (short)0, acc[2 * t + 1], false, false);
            }
            apLo += 32; apHi += 32; bp += 32;
        }

        #pragma unroll
        for (int t = 0; t < 7; ++t) {
            const int n = n0 + t * 16 + ml;
            const float bias = b2[n];
            #pragma unroll
            for (int mt = 0; mt < 2; ++mt) {
                #pragma unroll
                for (int i = 0; i < 8; ++i) {
                    const int mr = m0 + mt * 16 + i + 8 * kg;
                    const int t3 = tok[mr * 4 + 3];
                    out[(size_t)mr * DD + n] =
                        acc[2 * t + mt][i] + bias + table32[(size_t)t3 * DD + n];
                }
            }
        }
    }
}

// ---------------- launch ----------------

extern "C" void kernel_launch(void* const* d_in, const int* in_sizes, int n_in,
                              void* d_out, int out_size, void* d_ws, size_t ws_size,
                              hipStream_t stream) {
    const int*   pf    = (const int*)  d_in[0];   // (B, 4L) int32
    const int*   plen  = (const int*)  d_in[1];   // (B,)    int32
    const float* table = (const float*)d_in[2];   // (V, D)
    const float* W1    = (const float*)d_in[3];   // (3D, D)
    const float* b1    = (const float*)d_in[4];   // (D,)
    const float* W2    = (const float*)d_in[5];   // (D, D)
    const float* b2    = (const float*)d_in[6];   // (D,)

    float* out      = (float*)d_out;              // [MM*DD] out, then [MM] mask
    float* mask_out = out + (size_t)MM * DD;

    // workspace layout (bytes), 256-aligned
    char* ws = (char*)d_ws;
    size_t off = 0;
    __bf16* tableb = (__bf16*)(ws + off); off += (size_t)VV * DD * 2;
    off = (off + 255) & ~(size_t)255;
    __bf16* W1t    = (__bf16*)(ws + off); off += (size_t)DD * K1 * 2;
    off = (off + 255) & ~(size_t)255;
    __bf16* W2t    = (__bf16*)(ws + off); off += (size_t)DD * K2 * 2;
    off = (off + 255) & ~(size_t)255;
    int*    tokbuf = (int*)(ws + off);    off += (size_t)MM * 4 * 4;

    // prep
    {
        int n = VV * DD;
        cvt_bf16_kernel<<<(n + 255) / 256, 256, 0, stream>>>(table, tableb, n);
    }
    {
        int n = K1 * DD;
        transpose_cvt_kernel<<<(n + 255) / 256, 256, 0, stream>>>(W1, W1t, K1, DD);
    }
    {
        int n = K2 * DD;
        transpose_cvt_kernel<<<(n + 255) / 256, 256, 0, stream>>>(W2, W2t, K2, DD);
    }
    tok_mask_kernel<<<(MM + 255) / 256, 256, 0, stream>>>(pf, plen, tokbuf, mask_out);

    // fused gather + GEMM1 + GELU + GEMM2 + residual (h lives in LDS)
    fused_mlp_kernel<<<MM / 32, 256, 0, stream>>>(tableb, W1t, W2t, tokbuf,
                                                  b1, b2, table, out);
}
